// ModalityAttention_89850715832388
// MI455X (gfx1250) — matrix operand order repaired and checked by reading, not compiled, not run on hardware
//
#include <hip/hip_runtime.h>

#define Bc 4
#define Tc 64
#define Nc 16
#define Mc 8
#define Dc 512

typedef __attribute__((ext_vector_type(16))) __bf16 v16bf;
typedef __attribute__((ext_vector_type(8)))  float  v8f;
typedef __attribute__((ext_vector_type(4)))  unsigned int u32x4;

union Frag { v16bf v; u32x4 q[2]; };

__device__ __forceinline__ unsigned short f2bf(float f) {
  unsigned u = __float_as_uint(f);
  u += 0x7fffu + ((u >> 16) & 1u);      // round-to-nearest-even
  return (unsigned short)(u >> 16);
}

// ---------------- kernel 0: fp32 -> bf16 weights (3 x 512x512) ----------------
__global__ void cvt_w_kernel(const float* __restrict__ wq,
                             const float* __restrict__ wk,
                             const float* __restrict__ wv,
                             unsigned short* __restrict__ out) {
  int gid = blockIdx.x * 256 + threadIdx.x;     // 0..196607
  int eid = gid * 4;                            // 4 elements per thread
  const float* src = (eid < (1 << 18)) ? wq : (eid < (2 << 18)) ? wk : wv;
  int off = eid & ((1 << 18) - 1);
  float4 f = *(const float4*)(src + off);
  uint2 p;
  p.x = (unsigned)f2bf(f.x) | ((unsigned)f2bf(f.y) << 16);
  p.y = (unsigned)f2bf(f.z) | ((unsigned)f2bf(f.w) << 16);
  *(uint2*)(out + eid) = p;
}

// ---------------- fused modality-attention kernel, one WG per (t,n) ----------------
__global__ void ModalityAttention_kernel(const float* __restrict__ x,
                                         const unsigned short* __restrict__ wb, // 3*512*512 bf16
                                         const float* __restrict__ bq,
                                         const float* __restrict__ bk,
                                         const float* __restrict__ bv,
                                         float* __restrict__ out) {
  extern __shared__ char smem[];
  unsigned short* lds_x = (unsigned short*)smem;          // 32*512 bf16      (32 KB)
  unsigned short* lds_q = lds_x + 32 * Dc;                // 4*16*512 bf16    (64 KB, m padded)
  unsigned short* lds_k = lds_q + 4 * 16 * Dc;            // 4*16*512 bf16    (64 KB, m padded)
  float* lds_v = (float*)(smem + (32 * Dc + 2 * 4 * 16 * Dc) * 2); // 32*512 f32 (64 KB)
  float* lds_s = lds_v + 32 * Dc;                         // 8x8 scores
  float* lds_w = lds_s + 64;                              // 8x8 softmax weights

  const int tid  = threadIdx.x;
  const int wid  = tid >> 5;
  const int lane = tid & 31;
  const int lm   = lane & 15;
  const int lh   = lane >> 4;
  const int t    = blockIdx.x >> 4;
  const int n    = blockIdx.x & 15;

  // ---- zero Q/K LDS (pad rows m=8..15 must stay zero) ----
  {
    unsigned* z = (unsigned*)lds_q;
    #pragma unroll 1
    for (int i = tid; i < 2 * 4 * 16 * Dc / 2; i += 256) z[i] = 0u;
  }
  // ---- stage this WG's 32 x-rows to LDS as bf16 ----
  {
    int row = tid >> 3;                 // 0..31 -> (b, m)
    int bb = row >> 3, mm = row & 7;
    int d0 = (tid & 7) * 64;
    const float4* src = (const float4*)(x + ((((size_t)bb * Tc + t) * Nc + n) * Mc + mm) * Dc + d0);
    #pragma unroll
    for (int j = 0; j < 16; ++j) {
      float4 f = src[j];
      uint2 p;
      p.x = (unsigned)f2bf(f.x) | ((unsigned)f2bf(f.y) << 16);
      p.y = (unsigned)f2bf(f.z) | ((unsigned)f2bf(f.w) << 16);
      *(uint2*)(lds_x + row * Dc + d0 + j * 4) = p;
    }
  }
  __syncthreads();

  // ---- Q/K/V projections: per wave, a 32x64 slab of the 32x512 result ----
  const int colBase = wid * 64;
  #pragma unroll 1
  for (int proj = 0; proj < 3; ++proj) {
    const unsigned short* W = wb + proj * (Dc * Dc);      // row-major [e][d]; B = W^T
    const float* bias = (proj == 0) ? bq : (proj == 1) ? bk : bv;

    v8f acc[2][4] = {};
    #pragma unroll 1
    for (int kt = 0; kt < 16; ++kt) {
      int kk0 = kt * 32;
      __builtin_prefetch(W + (colBase + lm) * Dc + kk0 + 32, 0, 3);  // global_prefetch_b8
      Frag bfr[4];
      #pragma unroll
      for (int ct = 0; ct < 4; ++ct) {
        // B-frag: column e = W row, k-pairs contiguous along d
        int e = colBase + ct * 16 + lm;
        const u32x4* p = (const u32x4*)(W + e * Dc + kk0 + lh * 8);
        bfr[ct].q[0] = p[0];
        bfr[ct].q[1] = p[2];
      }
      #pragma unroll
      for (int rt = 0; rt < 2; ++rt) {
        Frag afr;                       // A-frag: row m = lm, K halves split by lh
        int base = (rt * 16 + lm) * Dc + kk0 + lh * 8;
        afr.q[0] = *(const u32x4*)(lds_x + base);
        afr.q[1] = *(const u32x4*)(lds_x + base + 16);
        #pragma unroll
        for (int ct = 0; ct < 4; ++ct) {
          acc[rt][ct] = __builtin_amdgcn_wmma_f32_16x16x32_bf16(
              false, afr.v, false, bfr[ct].v, (short)0, acc[rt][ct], false, false);
        }
      }
    }
    // ---- epilogue: +bias, scatter to LDS. proj branch hoisted (uniform) so the
    //      unrolled store loops are straight-line ds_store streams. ----
    // For fixed (rt,lh): rows r=0..7 stay in one (b,m)-octet:
    //   padded Q/K row = (2*rt+lh)*16 + r ;  V row = 8*(2*rt+lh) + r
    if (proj < 2) {
      unsigned short* dstqk = (proj == 0) ? lds_q : lds_k;
      #pragma unroll
      for (int ct = 0; ct < 4; ++ct) {
        int e = colBase + ct * 16 + lm;
        float bvv = bias[e];
        #pragma unroll
        for (int rt = 0; rt < 2; ++rt) {
          unsigned short* p0 = dstqk + ((2 * rt + lh) * 16) * Dc + e;
          #pragma unroll
          for (int r = 0; r < 8; ++r)
            p0[r * Dc] = f2bf(acc[rt][ct][r] + bvv);
        }
      }
    } else {
      #pragma unroll
      for (int ct = 0; ct < 4; ++ct) {
        int e = colBase + ct * 16 + lm;
        float bvv = bias[e];
        #pragma unroll
        for (int rt = 0; rt < 2; ++rt) {
          float* p0 = lds_v + (8 * (2 * rt + lh)) * Dc + e;
          #pragma unroll
          for (int r = 0; r < 8; ++r)
            p0[r * Dc] = acc[rt][ct][r] + bvv;
        }
      }
    }
  }
  __syncthreads();

  // ---- scores: S[m,k] = sum_{b,d} Q[b,m,d] K[b,k,d]  (K'=2048, 64 WMMA steps, wave 0) ----
  if (wid == 0) {
    v8f s = {};
    #pragma unroll 1
    for (int kt = 0; kt < 64; ++kt) {
      int kk = kt * 32;
      int bb = kk >> 9;
      int d0 = kk & 511;
      Frag a, b;
      int abase = (bb * 16 + lm) * Dc + d0 + lh * 8;      // A from Q (rows 8..15 zero-padded)
      a.q[0] = *(const u32x4*)(lds_q + abase);
      a.q[1] = *(const u32x4*)(lds_q + abase + 16);
      int bbase = (bb * 16 + lm) * Dc + d0 + lh * 8;      // B = Kcat^T: table-order d runs of K row
      b.q[0] = *(const u32x4*)(lds_k + bbase);
      b.q[1] = *(const u32x4*)(lds_k + bbase + 16);
      s = __builtin_amdgcn_wmma_f32_16x16x32_bf16(false, a.v, false, b.v, (short)0, s, false, false);
    }
    const float scale = 0.044194173824159216f;            // 1/sqrt(512)
    if (lane < 8) {                                        // lanes 0..7: cols k=0..7, rows r
      #pragma unroll
      for (int r = 0; r < 8; ++r) lds_s[r * 8 + lane] = s[r] * scale;
    }
    __asm__ volatile("s_wait_dscnt 0" ::: "memory");
    if (lane < 8) {                                        // per-row softmax over 8 values
      float v0[8], mx = -1e30f;
      #pragma unroll
      for (int k = 0; k < 8; ++k) { v0[k] = lds_s[lane * 8 + k]; mx = fmaxf(mx, v0[k]); }
      float sum = 0.f;
      #pragma unroll
      for (int k = 0; k < 8; ++k) { v0[k] = __expf(v0[k] - mx); sum += v0[k]; }
      float inv = 1.f / sum;
      #pragma unroll
      for (int k = 0; k < 8; ++k) lds_w[lane * 8 + k] = v0[k] * inv;
    }
  }
  __syncthreads();

  // ---- output: out[b,m,d] = sum_k w[m,k] * V[b,k,d] ----
  {
    int row = tid >> 3;
    int bb = row >> 3, mm = row & 7;
    int d0 = (tid & 7) * 64;
    float w[8];
    #pragma unroll
    for (int k = 0; k < 8; ++k) w[k] = lds_w[mm * 8 + k];
    float4* dst = (float4*)(out + ((((size_t)bb * Tc + t) * Nc + n) * Mc + mm) * Dc + d0);
    #pragma unroll 1
    for (int j = 0; j < 16; ++j) {
      int d = d0 + j * 4;
      float4 o = make_float4(0.f, 0.f, 0.f, 0.f);
      #pragma unroll
      for (int k = 0; k < 8; ++k) {
        const float* vr = lds_v + (bb * 8 + k) * Dc + d;
        o.x += w[k] * vr[0]; o.y += w[k] * vr[1];
        o.z += w[k] * vr[2]; o.w += w[k] * vr[3];
      }
      dst[j] = o;
    }
  }
}

extern "C" void kernel_launch(void* const* d_in, const int* in_sizes, int n_in,
                              void* d_out, int out_size, void* d_ws, size_t ws_size,
                              hipStream_t stream) {
  const float* x  = (const float*)d_in[0];
  const float* Wq = (const float*)d_in[1];
  const float* bq = (const float*)d_in[2];
  const float* Wk = (const float*)d_in[3];
  const float* bk = (const float*)d_in[4];
  const float* Wv = (const float*)d_in[5];
  const float* bv = (const float*)d_in[6];
  float* out = (float*)d_out;
  unsigned short* wsW = (unsigned short*)d_ws;   // 3*512*512 bf16 = 1.5 MB

  // weights fp32 -> bf16
  cvt_w_kernel<<<768, 256, 0, stream>>>(Wq, Wk, Wv, wsW);

  // fused attention: one WG per (t,n); ~225 KB dynamic LDS (< 320 KB/WGP)
  const int smemBytes = (32 * Dc + 2 * 4 * 16 * Dc) * 2 + (32 * Dc + 128) * 4;
  (void)hipFuncSetAttribute(reinterpret_cast<const void*>(ModalityAttention_kernel),
                            hipFuncAttributeMaxDynamicSharedMemorySize, smemBytes);
  ModalityAttention_kernel<<<Tc * Nc, 256, smemBytes, stream>>>(x, wsW, bq, bk, bv, out);
}
